// MessagePassingNet_73864847557249
// MI455X (gfx1250) — matrix operand order repaired and checked by reading, not compiled, not run on hardware
//
#include <hip/hip_runtime.h>
#include <hip/hip_bf16.h>

typedef __attribute__((ext_vector_type(16))) _Float16 v16h;
typedef __attribute__((ext_vector_type(8)))  _Float16 v8h;
typedef __attribute__((ext_vector_type(8)))  float    v8f;
typedef __attribute__((ext_vector_type(4)))  unsigned int v4u;
typedef __attribute__((ext_vector_type(8)))  unsigned int v8u;

#define NN   100000
#define NE   1200000
#define NG   1000
#define DIM  64
#define NDEG 11
#define WELEM (NDEG * DIM * DIM)          // 45056 elements per weight tensor
#define LDS_BYTES (2 * WELEM * 2 + NDEG * DIM * 4)   // Wl + Wr (f16) + bias (f32) = 183040

// ---------------------------------------------------------------------------
// WMMA wrapper: D = A(16x32 f16) * B(32x16 f16) + C(16x16 f32)
// ---------------------------------------------------------------------------
__device__ __forceinline__ v8f wmma_f16(v16h a, v16h b, v8f c) {
  return __builtin_amdgcn_wmma_f32_16x16x32_f16(
      /*neg_a=*/false, a, /*neg_b=*/false, b,
      /*c_mod=*/(short)0, c, /*reuse_a=*/false, /*reuse_b=*/false);
}

// Load one 16x32 f16 A-fragment for this lane from an f32 row (row-major, 64 wide).
// Lane L: row = base + (L&15); element e<8 -> K = 32*kb + 8*(L>>4) + e,
//         e>=8 -> K = 32*kb + 16 + 8*(L>>4) + (e-8).
__device__ __forceinline__ v16h loadA(const float* rowp, int kb, int half) {
  const float* p = rowp + kb * 32 + half * 8;
  v16h a;
#pragma unroll
  for (int i = 0; i < 8; ++i) a[i] = (_Float16)p[i];
#pragma unroll
  for (int i = 0; i < 8; ++i) a[8 + i] = (_Float16)p[16 + i];
  return a;
}

__device__ __forceinline__ v16h maskA(v16h a, bool keep) {
  v16h r;
#pragma unroll
  for (int i = 0; i < 16; ++i) r[i] = keep ? a[i] : (_Float16)0.0f;
  return r;
}

// B fragment from LDS weights stored as [d][n][k] f16: lane L -> col n = ct*16+(L&15),
// element e<8 -> K = 32*kb + 8*(L>>4) + e, e>=8 -> K = 32*kb + 16 + 8*(L>>4) + (e-8).
__device__ __forceinline__ v16h loadB(const _Float16* w, int kb) {
  const v8h* q = (const v8h*)(w + kb * 32);
  v8h lo = q[0], hi = q[2];
  v16h b;
#pragma unroll
  for (int i = 0; i < 8; ++i) { b[i] = lo[i]; b[8 + i] = hi[i]; }
  return b;
}

// ---------------------------------------------------------------------------
// Prep: transpose weight tensor [d][k][n] f32 -> [d][n][k] f16
// ---------------------------------------------------------------------------
__global__ void cvt_transpose_kernel(const float* __restrict__ src,
                                     _Float16* __restrict__ dst) {
  int i = blockIdx.x * blockDim.x + threadIdx.x;
  if (i >= WELEM) return;
  int d = i / (DIM * DIM);
  int r = i % (DIM * DIM);
  int k = r / DIM;
  int n = r % DIM;
  dst[d * DIM * DIM + n * DIM + k] = (_Float16)src[i];
}

// ---------------------------------------------------------------------------
// Edge scatter: h[dst] += feat[src]; optionally deg[dst] += 1
// One thread handles one edge x 4 features.
// ---------------------------------------------------------------------------
__global__ void scatter_kernel(const float* __restrict__ feat,
                               const int* __restrict__ ei,
                               float* __restrict__ hbuf,
                               int* __restrict__ deg, int with_deg) {
  int idx = blockIdx.x * blockDim.x + threadIdx.x;
  if (idx >= NE * 16) return;
  int e  = idx >> 4;
  int f4 = (idx & 15) * 4;
  __builtin_prefetch(ei + e + 4096, 0, 0);   // global_prefetch_b8 on gfx1250
  int src = ei[e];
  int dst = ei[NE + e];
  float4 v = *(const float4*)(feat + (size_t)src * DIM + f4);
  float* hp = hbuf + (size_t)dst * DIM + f4;
  atomicAdd(hp + 0, v.x);
  atomicAdd(hp + 1, v.y);
  atomicAdd(hp + 2, v.z);
  atomicAdd(hp + 3, v.w);
  if (with_deg && (idx & 15) == 0) atomicAdd(&deg[dst], 1);
}

// ---------------------------------------------------------------------------
// MFConv GEMM: out_i = Wl[deg_i]^T . h_i + Wr[deg_i]^T . x_i + bl[deg_i]
// One wave per 16-node tile. mode 0: relu -> yout. mode 1: raw -> yout (emb)
// and atomic relu-pool into gpool[batch].
// Weights (Wl+Wr f16, 180224 B contiguous) staged into LDS via one TDM
// tensor_load_to_lds descriptor issued by wave 0.
// ---------------------------------------------------------------------------
__global__ void mfconv_gemm_kernel(const float* __restrict__ hbuf,
                                   const float* __restrict__ xin,
                                   const _Float16* __restrict__ wlT,
                                   const float* __restrict__ bias,
                                   const int* __restrict__ deg,
                                   const int* __restrict__ batch,
                                   float* __restrict__ yout,
                                   float* __restrict__ gpool,
                                   int mode) {
  extern __shared__ __align__(32) char smem[];
  _Float16* ldsWl = (_Float16*)smem;                 // 45056 halfs
  _Float16* ldsWr = ldsWl + WELEM;                   // 45056 halfs (contiguous after Wl)
  float*    ldsB  = (float*)(ldsWr + WELEM);         // 704 floats

  // --- TDM bulk copy: Wl+Wr (2*WELEM halfs = 22528 x 8B) global -> LDS -------
  if (threadIdx.x < 32) {                            // wave 0 only (uniform per wave)
    const unsigned long long ga = (unsigned long long)wlT;
    const unsigned ldsoff = (unsigned)(unsigned long long)(void*)ldsWl;
    const unsigned NELEM8 = (2u * WELEM * 2u) / 8u;  // 22528 elements of 8 bytes
    v4u g0; v8u g1; v4u gz = {0u, 0u, 0u, 0u};
    g0[0] = 1u;                                      // count=1, no gather, user mode
    g0[1] = ldsoff;                                  // lds_addr (bytes)
    g0[2] = (unsigned)ga;                            // global_addr[31:0]
    g0[3] = (unsigned)((ga >> 32) & 0x01ffffffu) | 0x80000000u; // addr[56:32] | type=2
    g1[0] = 0x00030000u;                             // data_size=3 (8B), mask=0
    g1[1] = NELEM8 << 16;                            // tensor_dim0[15:0] @ bits 63:48
    g1[2] = 1u << 16;                                // tensor_dim0 hi=0; tensor_dim1=1
    g1[3] = NELEM8 << 16;                            // tensor_dim1 hi=0; tile_dim0
    g1[4] = 1u;                                      // tile_dim1=1, tile_dim2=0
    g1[5] = NELEM8;                                  // tensor_dim0_stride[31:0]
    g1[6] = 0u;
    g1[7] = 0u;
    asm volatile("tensor_load_to_lds %0, %1, %2, %3"
                 :: "s"(g0), "s"(g1), "s"(gz), "s"(gz) : "memory");
    __builtin_amdgcn_s_wait_tensorcnt(0);
  }
  for (int i = threadIdx.x; i < NDEG * DIM; i += blockDim.x) ldsB[i] = bias[i];
  __syncthreads();

  const int lane = threadIdx.x & 31;
  const int wave = threadIdx.x >> 5;
  const int tile = blockIdx.x * 8 + wave;
  const int base = tile * 16;
  if (base >= NN) return;                 // wave-uniform: EXEC stays all-ones inside

  const int half = lane >> 4;
  const int m    = base + (lane & 15);

  int myDeg = deg[m];
  if (myDeg > 10) myDeg = 10;
  const int myBatch = batch[m];

  const float* hrow = hbuf + (size_t)m * DIM;
  const float* xrow = xin  + (size_t)m * DIM;
  const v16h ah0 = loadA(hrow, 0, half), ah1 = loadA(hrow, 1, half);
  const v16h ax0 = loadA(xrow, 0, half), ax1 = loadA(xrow, 1, half);

  v8f acc[4];
#pragma unroll
  for (int ct = 0; ct < 4; ++ct) acc[ct] = (v8f){0.f, 0.f, 0.f, 0.f, 0.f, 0.f, 0.f, 0.f};

  for (int d = 0; d < NDEG; ++d) {
    if (__ballot(myDeg == d) == 0ull) continue;   // wave-uniform skip
    const bool keep = (myDeg == d);
    const v16h mh0 = maskA(ah0, keep), mh1 = maskA(ah1, keep);
    const v16h mx0 = maskA(ax0, keep), mx1 = maskA(ax1, keep);
#pragma unroll
    for (int ct = 0; ct < 4; ++ct) {
      const int n = ct * 16 + (lane & 15);
      const _Float16* blp = ldsWl + d * DIM * DIM + (size_t)n * DIM + half * 8;
      const _Float16* brp = ldsWr + d * DIM * DIM + (size_t)n * DIM + half * 8;
      acc[ct] = wmma_f16(mh0, loadB(blp, 0), acc[ct]);
      acc[ct] = wmma_f16(mh1, loadB(blp, 1), acc[ct]);
      acc[ct] = wmma_f16(mx0, loadB(brp, 0), acc[ct]);
      acc[ct] = wmma_f16(mx1, loadB(brp, 1), acc[ct]);
    }
  }

  // Epilogue. C layout: acc[ct][v] -> row = base + half*8 + v, col = ct*16 + (lane&15).
#pragma unroll
  for (int ct = 0; ct < 4; ++ct) {
    const int col = ct * 16 + (lane & 15);
#pragma unroll
    for (int v = 0; v < 8; ++v) {
      const int r  = base + half * 8 + v;
      const int dr = __shfl(myDeg, half * 8 + v, 32);
      const float val = acc[ct][v] + ldsB[dr * DIM + col];
      if (mode == 0) {
        yout[(size_t)r * DIM + col] = fmaxf(val, 0.f);
      } else {
        yout[(size_t)r * DIM + col] = val;   // emb (pre-relu)
        const int gb = __shfl(myBatch, half * 8 + v, 32);
        atomicAdd(&gpool[gb * DIM + col], fmaxf(val, 0.f));
      }
    }
  }
}

// ---------------------------------------------------------------------------
// Tiny dense head: out[r][o] = b[o] + sum_k in[r][k] * W[k][o]
// ---------------------------------------------------------------------------
__global__ void head_lin_kernel(const float* __restrict__ in,
                                const float* __restrict__ W,
                                const float* __restrict__ b,
                                float* __restrict__ out,
                                int rows, int kdim, int odim) {
  int i = blockIdx.x * blockDim.x + threadIdx.x;
  if (i >= rows * odim) return;
  int r = i / odim, o = i % odim;
  float s = b[o];
  for (int k = 0; k < kdim; ++k) s += in[r * kdim + k] * W[k * odim + o];
  out[i] = s;
}

// ---------------------------------------------------------------------------
extern "C" void kernel_launch(void* const* d_in, const int* in_sizes, int n_in,
                              void* d_out, int out_size, void* d_ws, size_t ws_size,
                              hipStream_t stream) {
  const float* x    = (const float*)d_in[0];
  const int*   ei   = (const int*)d_in[1];
  const int*   batc = (const int*)d_in[2];
  const float* Wl1  = (const float*)d_in[4];
  const float* bl1  = (const float*)d_in[5];
  const float* Wr1  = (const float*)d_in[6];
  const float* Wl2  = (const float*)d_in[7];
  const float* bl2  = (const float*)d_in[8];
  const float* Wr2  = (const float*)d_in[9];
  const float* W1   = (const float*)d_in[10];
  const float* b1   = (const float*)d_in[11];
  const float* W2   = (const float*)d_in[12];
  const float* b2   = (const float*)d_in[13];

  float* emb  = (float*)d_out;                         // [NN*64]
  float* pred = (float*)d_out + (size_t)NN * DIM;      // [NG*32]

  // Workspace layout
  auto alignup = [](size_t v) { return (v + 255) & ~(size_t)255; };
  char* ws = (char*)d_ws;
  size_t o_h   = 0;                          size_t sz_h = (size_t)NN * DIM * 4;
  size_t o_y   = alignup(o_h + sz_h);
  size_t o_deg = alignup(o_y + sz_h);
  size_t o_w   = alignup(o_deg + (size_t)NN * 4);
  size_t o_g   = alignup(o_w + (size_t)4 * WELEM * 2);
  size_t o_hid = alignup(o_g + (size_t)NG * DIM * 4);

  float*    hbuf = (float*)(ws + o_h);
  float*    y1   = (float*)(ws + o_y);
  int*      deg  = (int*)(ws + o_deg);
  _Float16* wl1T = (_Float16*)(ws + o_w);
  _Float16* wr1T = wl1T + WELEM;
  _Float16* wl2T = wr1T + WELEM;
  _Float16* wr2T = wl2T + WELEM;
  float*    g    = (float*)(ws + o_g);
  float*    hid  = (float*)(ws + o_hid);

  // Zero accumulators (every call: atomics accumulate into these)
  hipMemsetAsync(hbuf, 0, sz_h, stream);
  hipMemsetAsync(deg,  0, (size_t)NN * 4, stream);
  hipMemsetAsync(g,    0, (size_t)NG * DIM * 4, stream);

  // Weight transpose+convert to f16 [d][n][k]
  {
    int blk = 256, grd = (WELEM + blk - 1) / blk;
    cvt_transpose_kernel<<<grd, blk, 0, stream>>>(Wl1, wl1T);
    cvt_transpose_kernel<<<grd, blk, 0, stream>>>(Wr1, wr1T);
    cvt_transpose_kernel<<<grd, blk, 0, stream>>>(Wl2, wl2T);
    cvt_transpose_kernel<<<grd, blk, 0, stream>>>(Wr2, wr2T);
  }

  const int sc_blk = 256, sc_grd = (NE * 16) / sc_blk;
  const int cv_blk = 256, cv_grd = (NN / 16 + 7) / 8;   // 8 tiles (waves) per block

  // Conv 1: neighbor sum of x, then bucketed GEMM + relu -> y1
  scatter_kernel<<<sc_grd, sc_blk, 0, stream>>>(x, ei, hbuf, deg, 1);
  mfconv_gemm_kernel<<<cv_grd, cv_blk, LDS_BYTES, stream>>>(
      hbuf, x, wl1T, bl1, deg, batc, y1, nullptr, 0);

  // Conv 2: neighbor sum of y1, GEMM -> emb (d_out) + relu-pool into g
  hipMemsetAsync(hbuf, 0, sz_h, stream);
  scatter_kernel<<<sc_grd, sc_blk, 0, stream>>>(y1, ei, hbuf, deg, 0);
  mfconv_gemm_kernel<<<cv_grd, cv_blk, LDS_BYTES, stream>>>(
      hbuf, y1, wl2T, bl2, deg, batc, emb, g, 1);

  // Head: hid = g@W1 + b1 ; pred = hid@W2 + b2
  head_lin_kernel<<<(NG * DIM + 255) / 256, 256, 0, stream>>>(g, W1, b1, hid, NG, DIM, DIM);
  head_lin_kernel<<<(NG * 32 + 255) / 256, 256, 0, stream>>>(hid, W2, b2, pred, NG, DIM, 32);
}
